// PlanCostEstimatorFull_21552145891414
// MI455X (gfx1250) — matrix operand order repaired and checked by reading, not compiled, not run on hardware
//
#include <hip/hip_runtime.h>
#include <hip/hip_bf16.h>

typedef __attribute__((ext_vector_type(16))) _Float16 v16h;
typedef __attribute__((ext_vector_type(8)))  float    v8f;
typedef __attribute__((ext_vector_type(4)))  float    v4f;

union Frag { v4f f[2]; v16h h; };

#define NTHREADS 256
#define XSTRIDE  208   // halfs per activation row (node-major), C<=200 padded
#define BSTRIDE  608   // halfs per gathered row, Kmax=608
#define YSTRIDE  208   // floats per raw-output row

// ---- LDS carve (bytes) ----
static constexpr int OFF_XT   = 0;
static constexpr int OFF_BM   = OFF_XT   + 64 * XSTRIDE * 2;   // 26624
static constexpr int OFF_Y32  = OFF_BM   + 64 * BSTRIDE * 2;   // 104448
static constexpr int OFF_REDS = OFF_Y32  + 64 * YSTRIDE * 4;   // 157696
static constexpr int OFF_REDQ = OFF_REDS + NTHREADS * 4;
static constexpr int OFF_GP   = OFF_REDQ + NTHREADS * 4;
static constexpr int OFF_GV   = OFF_GP   + NTHREADS * 4;
static constexpr int OFF_COMB = OFF_GV   + 64 * 4;
static constexpr int OFF_Z1   = OFF_COMB + 200 * 4;
static constexpr int OFF_Z2   = OFF_Z1   + 128 * 4;
static constexpr int OFF_IDX  = OFF_Z2   + 64 * 4;
static constexpr int OFF_ST   = OFF_IDX  + 192 * 4;
static constexpr int SMEM_BYTES = OFF_ST + 16;

// Pack conv weights (Cout,Cin,3) f32 -> f16 [Mpad][Kpad], K = k*Cin + i, zero padded.
__global__ void pack_w_kernel(const float* __restrict__ w, _Float16* __restrict__ dst,
                              int Cin, int Cout, int Mpad, int Kpad) {
    int e = blockIdx.x * blockDim.x + threadIdx.x;
    if (e >= Mpad * Kpad) return;
    int o = e / Kpad, K = e - o * Kpad;
    float v = 0.f;
    if (o < Cout && K < 3 * Cin) {
        int k = K / Cin, i = K - k * Cin;
        v = w[(o * Cin + i) * 3 + k];
    }
    dst[e] = (_Float16)v;
}

// Build gathered B-matrix: Bm[m][k*C+i] = xT[idx[3m+k]][i]; zero K-tail and row 63.
__device__ __forceinline__ void build_B(const _Float16* xT, _Float16* Bm,
                                        const int* idxs, int C, int Kpad, int tid) {
    const int cpk = C >> 2;              // 8-byte chunks per (m,k) copy
    const int total = 189 * cpk;
    for (int e = tid; e < total; e += NTHREADS) {
        int p  = e / cpk;                // p = 3*m + k
        int c4 = e - p * cpk;
        int m  = p / 3;
        int k  = p - m * 3;
        int s  = idxs[p];
        *(unsigned long long*)(Bm + m * BSTRIDE + k * C + 4 * c4) =
            *(const unsigned long long*)(xT + s * XSTRIDE + 4 * c4);
    }
    const int tail = Kpad - 3 * C;
    for (int e = tid; e < 64 * tail; e += NTHREADS) {
        int m = e / tail, c = e - m * tail;
        Bm[m * BSTRIDE + 3 * C + c] = (_Float16)0.f;
    }
    for (int e = tid; e < Kpad; e += NTHREADS)
        Bm[63 * BSTRIDE + e] = (_Float16)0.f;
}

// Tiled WMMA GEMM with 4-way A-fragment reuse: each wave owns one 16-row M-tile
// and accumulates all four 16-col N-tiles (64 nodes) in registers.
// y32[m+1][o] = sum_K Wp[o][K]*Bm[m][K] + bias[o]
__device__ __forceinline__ void gemm_wmma(const _Float16* __restrict__ Wp,
                                          const float* __restrict__ bias,
                                          const _Float16* Bm, float* y32,
                                          int Cout, int Kpad, int tid) {
    const int Mtiles = (Cout + 15) >> 4;
    const int wave   = tid >> 5;
    const int lane   = tid & 31;
    const int l15    = lane & 15;
    const int hi     = lane >> 4;
    for (int mt = wave; mt < Mtiles; mt += 8) {  // wave-uniform: EXEC all-ones at WMMA
        v8f acc[4];
        #pragma unroll
        for (int nt = 0; nt < 4; ++nt) acc[nt] = (v8f){0.f,0.f,0.f,0.f,0.f,0.f,0.f,0.f};
        // A frag: row M = mt*16+l15; K runs [8*hi, 8*hi+8) and [8*hi+16, 8*hi+24)
        const _Float16* aP = Wp + (size_t)(mt * 16 + l15) * Kpad + 8 * hi;
        // B frag base (nt=0): col N = l15; K runs [8*hi, 8*hi+8) and [8*hi+16, 8*hi+24).
        // N-tile stride is a constant 16*BSTRIDE halfs -> folds into ds offsets.
        const _Float16* bP = Bm + l15 * BSTRIDE + 8 * hi;
        for (int kc = 0; kc < Kpad; kc += 32) {
            Frag Fa;
            Fa.f[0] = *(const v4f*)(aP + kc);
            Fa.f[1] = *(const v4f*)(aP + kc + 16);
            #pragma unroll
            for (int nt = 0; nt < 4; ++nt) {
                Frag Fb;
                const _Float16* bn = bP + nt * (16 * BSTRIDE) + kc;
                Fb.f[0] = *(const v4f*)(bn);
                Fb.f[1] = *(const v4f*)(bn + 16);
                acc[nt] = __builtin_amdgcn_wmma_f32_16x16x32_f16(
                    false, Fa.h, false, Fb.h, (short)0, acc[nt], false, false);
            }
        }
        const int ob = mt * 16 + 8 * hi;         // C/D layout: elem r -> M = r + 8*hi
        #pragma unroll
        for (int nt = 0; nt < 4; ++nt) {
            const int m = nt * 16 + l15;         // output column (node m+1)
            if (m < 63) {
                #pragma unroll
                for (int r = 0; r < 8; ++r) {
                    const int o = ob + r;
                    if (o < Cout) y32[(m + 1) * YSTRIDE + o] = acc[nt][r] + bias[o];
                }
            }
        }
    }
}

// Layernorm over all 64*C elements (ddof=1) + ReLU, write f16 activations.
__device__ __forceinline__ void layernorm_relu(const float* y32, _Float16* xT, int C,
                                               int tid, float* redS, float* redQ,
                                               float* stats) {
    const int nelem = 64 * C;
    float s = 0.f, q = 0.f;
    for (int e = tid; e < nelem; e += NTHREADS) {
        int n = e / C, c = e - n * C;
        float v = y32[n * YSTRIDE + c];
        s += v; q += v * v;
    }
    redS[tid] = s; redQ[tid] = q;
    __syncthreads();
    for (int off = NTHREADS / 2; off > 0; off >>= 1) {
        if (tid < off) { redS[tid] += redS[tid + off]; redQ[tid] += redQ[tid + off]; }
        __syncthreads();
    }
    if (tid == 0) {
        float n  = (float)nelem;
        float mu = redS[0] / n;
        float var = (redQ[0] - n * mu * mu) / (n - 1.f);
        float sd = sqrtf(fmaxf(var, 0.f));
        stats[0] = mu;
        stats[1] = 1.f / (sd + 1e-5f);
    }
    __syncthreads();
    const float mu = stats[0], inv = stats[1];
    for (int e = tid; e < nelem; e += NTHREADS) {
        int n = e / C, c = e - n * C;
        xT[n * XSTRIDE + c] =
            (_Float16)fmaxf(0.f, (y32[n * YSTRIDE + c] - mu) * inv);
    }
}

__global__ __launch_bounds__(NTHREADS)
void plan_cost_kernel(const float* __restrict__ trees,
                      const int* __restrict__ indexes,
                      const unsigned char* __restrict__ maskp,
                      const float* __restrict__ c1b, const float* __restrict__ c2b,
                      const float* __restrict__ c3b,
                      const float* __restrict__ gw1, const float* __restrict__ gb1,
                      const float* __restrict__ gw2, const float* __restrict__ gb2,
                      const float* __restrict__ rw1, const float* __restrict__ rb1,
                      const float* __restrict__ rw2, const float* __restrict__ rb2,
                      const float* __restrict__ rw3, const float* __restrict__ rb3,
                      const _Float16* __restrict__ W1p,
                      const _Float16* __restrict__ W2p,
                      const _Float16* __restrict__ W3p,
                      float* __restrict__ out) {
    extern __shared__ char smem[];
    _Float16* xT   = (_Float16*)(smem + OFF_XT);
    _Float16* Bm   = (_Float16*)(smem + OFF_BM);
    float*    y32  = (float*)(smem + OFF_Y32);
    float*    redS = (float*)(smem + OFF_REDS);
    float*    redQ = (float*)(smem + OFF_REDQ);
    float*    gp   = (float*)(smem + OFF_GP);
    float*    gv   = (float*)(smem + OFF_GV);
    float*    comb = (float*)(smem + OFF_COMB);
    float*    z1   = (float*)(smem + OFF_Z1);
    float*    z2   = (float*)(smem + OFF_Z2);
    int*      idxs = (int*)(smem + OFF_IDX);
    float*    st   = (float*)(smem + OFF_ST);

    const int b = blockIdx.x;
    const int tid = threadIdx.x;

    // stage inputs: indexes + transposed f16 activations
    for (int e = tid; e < 189; e += NTHREADS) idxs[e] = indexes[b * 189 + e];
    for (int e = tid; e < 200 * 64; e += NTHREADS) {
        int i = e >> 6, n = e & 63;
        xT[n * XSTRIDE + i] = (_Float16)trees[(size_t)(b * 200 + i) * 64 + n];
    }
    __syncthreads();

    // ---- layer 1: C=200 -> 200 ----
    build_B(xT, Bm, idxs, 200, 608, tid);
    for (int c = tid; c < 200; c += NTHREADS) y32[c] = 0.f;   // node-0 column
    __syncthreads();
    gemm_wmma(W1p, c1b, Bm, y32, 200, 608, tid);
    __syncthreads();
    layernorm_relu(y32, xT, 200, tid, redS, redQ, st);
    __syncthreads();

    // ---- layer 2: C=200 -> 100 ----
    build_B(xT, Bm, idxs, 200, 608, tid);
    for (int c = tid; c < 100; c += NTHREADS) y32[c] = 0.f;
    __syncthreads();
    gemm_wmma(W2p, c2b, Bm, y32, 100, 608, tid);
    __syncthreads();
    layernorm_relu(y32, xT, 100, tid, redS, redQ, st);
    __syncthreads();

    // ---- layer 3: C=100 -> 100 (no LN) ----
    build_B(xT, Bm, idxs, 100, 320, tid);
    for (int c = tid; c < 100; c += NTHREADS) y32[c] = 0.f;
    __syncthreads();
    gemm_wmma(W3p, c3b, Bm, y32, 100, 320, tid);
    __syncthreads();
    // y32[n][f] == emb[b, n, f], n=0..63, f=0..99

    // ---- gate MLP: 4 threads per node over 50 hidden ----
    {
        const int n = tid >> 2, q = tid & 3;
        float part = 0.f;
        for (int h = q; h < 50; h += 4) {
            float d = gb1[h];
            for (int f = 0; f < 100; ++f) d += y32[n * YSTRIDE + f] * gw1[h * 100 + f];
            part += fmaxf(d, 0.f) * gw2[h];
        }
        gp[tid] = part;
    }
    __syncthreads();
    if ((tid & 3) == 0) {
        const int n = tid >> 2;
        float g = gp[tid] + gp[tid + 1] + gp[tid + 2] + gp[tid + 3] + gb2[0];
        if (maskp[b * 64 + n]) g = -__builtin_inff();
        gv[n] = g;
    }
    __syncthreads();
    if (tid == 0) {                         // softmax over 64 nodes
        float mx = gv[0];
        for (int n = 1; n < 64; ++n) mx = fmaxf(mx, gv[n]);
        float ssum = 0.f;
        for (int n = 0; n < 64; ++n) { float e = __expf(gv[n] - mx); gv[n] = e; ssum += e; }
        st[2] = 1.f / ssum;
    }
    __syncthreads();
    if (tid < 100) {
        const float invs = st[2];
        float p = 0.f;
        for (int n = 0; n < 64; ++n) p += gv[n] * y32[n * YSTRIDE + tid];
        comb[100 + tid] = p * invs;          // pooled
        comb[tid]       = y32[1 * YSTRIDE + tid];  // root = emb[:,1,:]
    }
    __syncthreads();
    if (tid < 128) {
        float d = rb1[tid];
        for (int j = 0; j < 200; ++j) d += comb[j] * rw1[tid * 200 + j];
        z1[tid] = fmaxf(d, 0.f);
    }
    __syncthreads();
    if (tid < 64) {
        float d = rb2[tid];
        for (int j = 0; j < 128; ++j) d += z1[j] * rw2[tid * 128 + j];
        z2[tid] = fmaxf(d, 0.f);
    }
    __syncthreads();
    if (tid == 0) {
        float d = rb3[0];
        for (int j = 0; j < 64; ++j) d += z2[j] * rw3[j];
        out[b] = d;                          // tuple elem 0: (B,1)
    }
    for (int j = tid; j < 200; j += NTHREADS)
        out[2048 + (size_t)b * 200 + j] = comb[j];   // tuple elem 1: (B,200)
}

extern "C" void kernel_launch(void* const* d_in, const int* in_sizes, int n_in,
                              void* d_out, int out_size, void* d_ws, size_t ws_size,
                              hipStream_t stream) {
    const float* trees   = (const float*)d_in[0];
    const int*   indexes = (const int*)d_in[1];
    const unsigned char* maskp = (const unsigned char*)d_in[2];
    const float* c1w = (const float*)d_in[3];
    const float* c1b = (const float*)d_in[4];
    const float* c2w = (const float*)d_in[5];
    const float* c2b = (const float*)d_in[6];
    const float* c3w = (const float*)d_in[7];
    const float* c3b = (const float*)d_in[8];
    const float* gw1 = (const float*)d_in[9];
    const float* gb1 = (const float*)d_in[10];
    const float* gw2 = (const float*)d_in[11];
    const float* gb2 = (const float*)d_in[12];
    const float* rw1 = (const float*)d_in[13];
    const float* rb1 = (const float*)d_in[14];
    const float* rw2 = (const float*)d_in[15];
    const float* rb2 = (const float*)d_in[16];
    const float* rw3 = (const float*)d_in[17];
    const float* rb3 = (const float*)d_in[18];

    // packed f16 weights in workspace
    _Float16* W1p = (_Float16*)d_ws;              // 208*608
    _Float16* W2p = W1p + 208 * 608;              // 112*608
    _Float16* W3p = W2p + 112 * 608;              // 112*320

    pack_w_kernel<<<(208 * 608 + 255) / 256, 256, 0, stream>>>(c1w, W1p, 200, 200, 208, 608);
    pack_w_kernel<<<(112 * 608 + 255) / 256, 256, 0, stream>>>(c2w, W2p, 200, 100, 112, 608);
    pack_w_kernel<<<(112 * 320 + 255) / 256, 256, 0, stream>>>(c3w, W3p, 100, 100, 112, 320);

    (void)hipFuncSetAttribute((const void*)plan_cost_kernel,
                              hipFuncAttributeMaxDynamicSharedMemorySize, SMEM_BYTES);

    plan_cost_kernel<<<dim3(2048), dim3(NTHREADS), SMEM_BYTES, stream>>>(
        trees, indexes, maskp, c1b, c2b, c3b,
        gw1, gb1, gw2, gb2, rw1, rb1, rw2, rb2, rw3, rb3,
        W1p, W2p, W3p, (float*)d_out);
}
